// BayesianLinear_8169027797260
// MI455X (gfx1250) — compile-verified, matrix operand order checked by reading, not run on hardware
//
#include <hip/hip_runtime.h>
#include <stdint.h>

#define IN_F   4096
#define OUT_F  4096
#define TOKENS 4096

// ---- GEMM tiling ----
#define BM 128          // tokens per block
#define BN 256          // out-features per block
#define KC 32           // K per stage (one bf16 WMMA depth)
#define LDSROW 80       // bytes per LDS tile row: 32 bf16 = 64B data + 16B pad (bank-conflict free)
#define A_HI_OFF 0
#define A_LO_OFF (BM * LDSROW)                       // 10240
#define B_HI_OFF (2 * BM * LDSROW)                   // 20480
#define B_LO_OFF (2 * BM * LDSROW + BN * LDSROW)     // 40960
#define BUF_BYTES (2 * BM * LDSROW + 2 * BN * LDSROW) // 61440; x2 buffers = 120KB LDS

typedef __attribute__((ext_vector_type(16))) __bf16 v16bf;
typedef __attribute__((ext_vector_type(8)))  float  v8f;
typedef __attribute__((ext_vector_type(4)))  float  v4f;
typedef __attribute__((ext_vector_type(4)))  unsigned int v4u;
typedef __attribute__((ext_vector_type(2)))  unsigned int v2u;

typedef __attribute__((address_space(3))) const v4u* lds_cv4u;

// Async-LDS builtin takes pointers to a gcc-vector of 4 ints (per hipcc diagnostic):
typedef int v4i_g __attribute__((__vector_size__(16)));
typedef __attribute__((address_space(1))) v4i_g* gptr128_t;  // global src
typedef __attribute__((address_space(3))) v4i_g* lptr128_t;  // LDS dst

#if __has_builtin(__builtin_amdgcn_global_load_async_to_lds_b128)
#define HAVE_ASYNC_LDS 1
#else
#define HAVE_ASYNC_LDS 0
#endif

// ---------------- helpers ----------------

__device__ __forceinline__ float softplusf_dev(float r) {
  // numerically stable softplus
  return fmaxf(r, 0.0f) + log1pf(expf(-fabsf(r)));
}

__device__ __forceinline__ unsigned f2bf_rne(float f) {
  unsigned u = __float_as_uint(f);
  return (u + 0x7FFFu + ((u >> 16) & 1u)) >> 16;   // round-to-nearest-even bf16
}
__device__ __forceinline__ float bf2f(unsigned h) { return __uint_as_float(h << 16); }

// 16B global -> LDS copy; async tensor-path copy when the builtin exists.
__device__ __forceinline__ void cp16(const void* g, void* l) {
#if HAVE_ASYNC_LDS
  __builtin_amdgcn_global_load_async_to_lds_b128(
      (gptr128_t)(size_t)g,
      (lptr128_t)(unsigned)(size_t)l, 0, 0);
#else
  *(v4u*)l = *(const v4u*)g;
#endif
}

__device__ __forceinline__ void wait_async_lds() {
#if HAVE_ASYNC_LDS
#if __has_builtin(__builtin_amdgcn_s_wait_asynccnt)
  __builtin_amdgcn_s_wait_asynccnt(0);
#else
  asm volatile("s_wait_asynccnt 0x0" ::: "memory");
#endif
#endif
}

// Load one 16x(32 K) bf16 WMMA fragment from an LDS tile row block.
// lane m (0..15): K 0-7 at +0  and K 16-23 at +32
// lane m+16    : K 8-15 at +16 and K 24-31 at +48   (offset baked in by caller via kchunk)
__device__ __forceinline__ v16bf ld_frag(const char* p) {
  unsigned a = (unsigned)(size_t)p;   // low 32 bits of flat shared addr == LDS offset
  union { v4u u[2]; v16bf v; } t;
  t.u[0] = *(lds_cv4u)a;
  t.u[1] = *(lds_cv4u)(a + 32u);
  return t.v;
}

__device__ __forceinline__ v8f wmma_bf16(v16bf a, v16bf b, v8f c) {
  return __builtin_amdgcn_wmma_f32_16x16x32_bf16(false, a, false, b, (short)0, c,
                                                 false, false);
}

// ---------------- prep kernels ----------------

__global__ __launch_bounds__(256) void prep_weight_kernel(
    const float* __restrict__ mu, const float* __restrict__ rho,
    const float* __restrict__ eps, unsigned short* __restrict__ hi,
    unsigned short* __restrict__ lo, int use_lo, size_t n) {
  size_t i = ((size_t)blockIdx.x * 256 + threadIdx.x) * 4;
  if (i >= n) return;
  v4f m = *(const v4f*)(mu + i);
  v4f r = *(const v4f*)(rho + i);
  v4f e = *(const v4f*)(eps + i);
  unsigned h[4], l[4];
#pragma unroll
  for (int j = 0; j < 4; ++j) {
    float wv = fmaf(softplusf_dev(r[j]), e[j], m[j]);
    h[j] = f2bf_rne(wv);
    l[j] = use_lo ? f2bf_rne(wv - bf2f(h[j])) : 0u;
  }
  v2u ph = {h[0] | (h[1] << 16), h[2] | (h[3] << 16)};
  *(v2u*)(hi + i) = ph;
  if (use_lo) {
    v2u pl = {l[0] | (l[1] << 16), l[2] | (l[3] << 16)};
    *(v2u*)(lo + i) = pl;
  }
}

__global__ __launch_bounds__(256) void prep_split_kernel(
    const float* __restrict__ src, unsigned short* __restrict__ hi,
    unsigned short* __restrict__ lo, int use_lo, size_t n) {
  size_t i = ((size_t)blockIdx.x * 256 + threadIdx.x) * 4;
  if (i >= n) return;
  v4f s = *(const v4f*)(src + i);
  unsigned h[4], l[4];
#pragma unroll
  for (int j = 0; j < 4; ++j) {
    h[j] = f2bf_rne(s[j]);
    l[j] = use_lo ? f2bf_rne(s[j] - bf2f(h[j])) : 0u;
  }
  v2u ph = {h[0] | (h[1] << 16), h[2] | (h[3] << 16)};
  *(v2u*)(hi + i) = ph;
  if (use_lo) {
    v2u pl = {l[0] | (l[1] << 16), l[2] | (l[3] << 16)};
    *(v2u*)(lo + i) = pl;
  }
}

__global__ __launch_bounds__(256) void prep_bias_kernel(
    const float* __restrict__ mu, const float* __restrict__ rho,
    const float* __restrict__ eps, float* __restrict__ b, int n) {
  int i = blockIdx.x * 256 + threadIdx.x;
  if (i < n) b[i] = fmaf(softplusf_dev(rho[i]), eps[i], mu[i]);
}

// ---------------- GEMM ----------------

__device__ __forceinline__ void stage_tiles(
    const unsigned short* __restrict__ Xhi, const unsigned short* __restrict__ Xlo,
    const unsigned short* __restrict__ Whi, const unsigned short* __restrict__ Wlo,
    char* s, int m0, int n0, int kk, int tid, int use_lo) {
  // A tile: BM x KC bf16 = 512 x 16B chunks; 2 per thread (per array)
#pragma unroll
  for (int i = 0; i < 2; ++i) {
    int c = tid + i * 256;
    int row = c >> 2, co = c & 3;
    size_t goff = (size_t)(m0 + row) * IN_F + kk + co * 8;
    int loff = row * LDSROW + co * 16;
    cp16(Xhi + goff, s + A_HI_OFF + loff);
    if (use_lo) cp16(Xlo + goff, s + A_LO_OFF + loff);
  }
  // B tile: BN x KC bf16 = 1024 x 16B chunks; 4 per thread (per array)
#pragma unroll
  for (int i = 0; i < 4; ++i) {
    int c = tid + i * 256;
    int row = c >> 2, co = c & 3;
    size_t goff = (size_t)(n0 + row) * IN_F + kk + co * 8;
    int loff = row * LDSROW + co * 16;
    cp16(Whi + goff, s + B_HI_OFF + loff);
    if (use_lo) cp16(Wlo + goff, s + B_LO_OFF + loff);
  }
}

__global__ __launch_bounds__(256) void bayes_gemm_wmma(
    const unsigned short* __restrict__ Xhi, const unsigned short* __restrict__ Xlo,
    const unsigned short* __restrict__ Whi, const unsigned short* __restrict__ Wlo,
    const float* __restrict__ bias, float* __restrict__ out, int use_lo) {
  __shared__ char smem[2 * BUF_BYTES];

  const int tid  = threadIdx.x;
  const int lane = tid & 31;
  const int w    = tid >> 5;
  const int wm   = w >> 2;  // 0..1  -> 64-row slice of BM
  const int wn   = w & 3;   // 0..3  -> 64-col slice of BN
  const int m0   = blockIdx.y * BM;
  const int n0   = blockIdx.x * BN;

  v8f acc[4][4];
#pragma unroll
  for (int i = 0; i < 4; ++i)
#pragma unroll
    for (int j = 0; j < 4; ++j) acc[i][j] = {};

  const int r       = lane & 15;
  const int kchunk  = (lane >> 4) * 16;
  const int aRowB   = wm * 64;
  const int bRowB   = wn * 64;

  stage_tiles(Xhi, Xlo, Whi, Wlo, smem, m0, n0, 0, tid, use_lo);

  const int nk = IN_F / KC;
  for (int k = 0; k < nk; ++k) {
    wait_async_lds();
    __syncthreads();
    char* cur = smem + (k & 1) * BUF_BYTES;
    char* nxt = smem + ((k + 1) & 1) * BUF_BYTES;
    if (k + 1 < nk)
      stage_tiles(Xhi, Xlo, Whi, Wlo, nxt, m0, n0, (k + 1) * KC, tid, use_lo);

    v16bf ah[4], bh[4], al[4], bl[4];
#pragma unroll
    for (int t = 0; t < 4; ++t) {
      ah[t] = ld_frag(cur + A_HI_OFF + (aRowB + t * 16 + r) * LDSROW + kchunk);
      bh[t] = ld_frag(cur + B_HI_OFF + (bRowB + t * 16 + r) * LDSROW + kchunk);
    }
    if (use_lo) {
#pragma unroll
      for (int t = 0; t < 4; ++t) {
        al[t] = ld_frag(cur + A_LO_OFF + (aRowB + t * 16 + r) * LDSROW + kchunk);
        bl[t] = ld_frag(cur + B_LO_OFF + (bRowB + t * 16 + r) * LDSROW + kchunk);
      }
    }
#pragma unroll
    for (int mt = 0; mt < 4; ++mt)
#pragma unroll
      for (int nt = 0; nt < 4; ++nt)
        acc[mt][nt] = wmma_bf16(ah[mt], bh[nt], acc[mt][nt]);
    if (use_lo) {
#pragma unroll
      for (int mt = 0; mt < 4; ++mt)
#pragma unroll
        for (int nt = 0; nt < 4; ++nt) {
          acc[mt][nt] = wmma_bf16(ah[mt], bl[nt], acc[mt][nt]);
          acc[mt][nt] = wmma_bf16(al[mt], bh[nt], acc[mt][nt]);
        }
    }
    __syncthreads();
  }

  // C/D layout: element rr of v8f -> M = rr + (lane>=16 ? 8 : 0); N = lane & 15
  const int cn    = lane & 15;
  const int mhalf = (lane >> 4) * 8;
#pragma unroll
  for (int nt = 0; nt < 4; ++nt) {
    int col = n0 + wn * 64 + nt * 16 + cn;
    float bv = bias[col];
#pragma unroll
    for (int mt = 0; mt < 4; ++mt) {
      int rowb = m0 + wm * 64 + mt * 16 + mhalf;
#pragma unroll
      for (int rr = 0; rr < 8; ++rr)
        out[(size_t)(rowb + rr) * OUT_F + col] = acc[mt][nt][rr] + bv;
    }
  }
}

// ---------------- last-resort fallback (tiny workspace) ----------------

__global__ __launch_bounds__(256) void bayes_naive(
    const float* __restrict__ x, const float* __restrict__ wmu,
    const float* __restrict__ wrho, const float* __restrict__ weps,
    const float* __restrict__ bmu, const float* __restrict__ brho,
    const float* __restrict__ beps, float* __restrict__ out) {
  int o = blockIdx.x * 256 + threadIdx.x;
  int t = blockIdx.y;
  if (o >= OUT_F) return;
  const float* xr = x + (size_t)t * IN_F;
  const float* mr = wmu + (size_t)o * IN_F;
  const float* rr = wrho + (size_t)o * IN_F;
  const float* er = weps + (size_t)o * IN_F;
  float acc = 0.f;
  for (int i = 0; i < IN_F; ++i)
    acc = fmaf(xr[i], fmaf(softplusf_dev(rr[i]), er[i], mr[i]), acc);
  out[(size_t)t * OUT_F + o] = acc + fmaf(softplusf_dev(brho[o]), beps[o], bmu[o]);
}

// ---------------- host entry ----------------

extern "C" void kernel_launch(void* const* d_in, const int* in_sizes, int n_in,
                              void* d_out, int out_size, void* d_ws, size_t ws_size,
                              hipStream_t stream) {
  (void)in_sizes; (void)n_in; (void)out_size;
  const float* x    = (const float*)d_in[0];
  const float* wmu  = (const float*)d_in[1];
  const float* wrho = (const float*)d_in[2];
  const float* bmu  = (const float*)d_in[3];
  const float* brho = (const float*)d_in[4];
  const float* weps = (const float*)d_in[5];
  const float* beps = (const float*)d_in[6];
  float* out = (float*)d_out;

  const size_t nW = (size_t)OUT_F * IN_F;
  const size_t nX = (size_t)TOKENS * IN_F;
  const size_t biasBytes = OUT_F * sizeof(float);
  const size_t needSplit = 2 * (nW + nX) * 2 + biasBytes;  // hi+lo bf16 + bias
  const size_t needHi    = (nW + nX) * 2 + biasBytes;      // hi-only bf16 + bias

  char* ws = (char*)d_ws;
  int use_lo;
  unsigned short *Whi, *Wlo, *Xhi, *Xlo;
  float* biasv;

  if (ws_size >= needSplit) {
    use_lo = 1;
    Whi = (unsigned short*)ws;
    Wlo = (unsigned short*)(ws + nW * 2);
    Xhi = (unsigned short*)(ws + nW * 4);
    Xlo = (unsigned short*)(ws + nW * 4 + nX * 2);
    biasv = (float*)(ws + nW * 4 + nX * 4);
  } else if (ws_size >= needHi) {
    use_lo = 0;
    Whi = (unsigned short*)ws;            Wlo = Whi;
    Xhi = (unsigned short*)(ws + nW * 2); Xlo = Xhi;
    biasv = (float*)(ws + nW * 2 + nX * 2);
  } else {
    dim3 g(OUT_F / 256, TOKENS);
    bayes_naive<<<g, 256, 0, stream>>>(x, wmu, wrho, weps, bmu, brho, beps, out);
    return;
  }

  prep_weight_kernel<<<(unsigned)(nW / 1024), 256, 0, stream>>>(wmu, wrho, weps, Whi, Wlo, use_lo, nW);
  prep_split_kernel<<<(unsigned)(nX / 1024), 256, 0, stream>>>(x, Xhi, Xlo, use_lo, nX);
  prep_bias_kernel<<<OUT_F / 256, 256, 0, stream>>>(bmu, brho, beps, biasv, OUT_F);

  dim3 g(OUT_F / BN, TOKENS / BM);  // (16, 32)
  bayes_gemm_wmma<<<g, 256, 0, stream>>>(Xhi, Xlo, Whi, Wlo, biasv, out, use_lo);
}
